// NTM_12137577578569
// MI455X (gfx1250) — matrix operand order repaired and checked by reading, not compiled, not run on hardware
//
#include <hip/hip_runtime.h>

// ---------------------------------------------------------------------------
// NTM forward for MI455X (gfx1250): 16 independent workgroups, each owns a
// 16-row batch tile and runs all T=64 recurrent steps. GEMMs use
// v_wmma_f32_16x16x32_bf16; addressing/memory-update is VALU f32.
// ---------------------------------------------------------------------------

typedef __bf16 bf16_t;
typedef __bf16 v16bf __attribute__((ext_vector_type(16)));
typedef float  v8f   __attribute__((ext_vector_type(8)));

static constexpr int T_ = 64, B_ = 256, IN_ = 64, OUT_ = 64, H_ = 256, N_ = 128, C_ = 64;
static constexpr int LIN = IN_ + C_;     // 128 (LSTM input)
static constexpr int BT  = 16;           // batch rows per workgroup
static constexpr int NG  = 4 * H_;       // 1024 gate columns
static constexpr int PRW = 288;          // padded [pr(80) | pw(208)] width
static constexpr int KO  = H_ + C_;      // 320 (output GEMM K)

// workspace layout (bytes)
static constexpr size_t OFF_MEM  = 0;                          // B*N*C f32 = 8 MB
static constexpr size_t OFF_WIH  = 8388608;                    // 1024x128 bf16
static constexpr size_t OFF_WHH  = OFF_WIH  + 262144;          // 1024x256 bf16
static constexpr size_t OFF_WRW  = OFF_WHH  + 524288;          // 288x256 bf16 (padded read|write)
static constexpr size_t OFF_WOUT = OFF_WRW  + 147456;          // 64x320 bf16
static constexpr size_t OFF_BG   = OFF_WOUT + 40960;           // 1024 f32  (b_ih + b_hh)
static constexpr size_t OFF_BRW  = OFF_BG   + 4096;            // 288 f32 padded
static constexpr size_t OFF_BOUT = OFF_BRW  + 1280;            // 64 f32

__device__ __forceinline__ float sigm(float x)      { return 1.f / (1.f + __expf(-x)); }
__device__ __forceinline__ float softplusf(float x) { return x > 20.f ? x : log1pf(__expf(x)); }

__device__ __forceinline__ v8f wmma_bf16(v16bf a, v16bf b, v8f c) {
  return __builtin_amdgcn_wmma_f32_16x16x32_bf16(false, a, false, b, (short)0, c, false, false);
}

// A-fragment (16xK slice starting at k0) from an LDS row-major bf16 matrix.
// Layout: lanes 0-15 hold K k0..k0+15 contiguous; lanes 16-31 hold k0+16..k0+31.
__device__ __forceinline__ v16bf load_a_lds(const bf16_t* base, int stride, int k0, int lane) {
  int m = lane & 15, hi = lane >> 4;
  const bf16_t* p = base + m * stride + k0 + 16 * hi;
  union { v16bf v; uint4 u[2]; } f;
  f.u[0] = *reinterpret_cast<const uint4*>(p);
  f.u[1] = *reinterpret_cast<const uint4*>(p + 8);
  return f.v;
}

// B-fragment (K x 16) = transpose of weight rows [ncol0..ncol0+15], K cols k0..k0+31.
// Lanes 0-15 (N=lane) hold K k0..k0+15 contiguous along the weight row; lanes 16-31 hold +16.
__device__ __forceinline__ v16bf load_b_glob(const bf16_t* W, int K, int ncol0, int k0, int lane) {
  int n = lane & 15, hi = lane >> 4;
  const bf16_t* p = W + (size_t)(ncol0 + n) * K + k0 + 16 * hi;
  union { v16bf v; uint4 u[2]; } f;
  f.u[0] = *reinterpret_cast<const uint4*>(p);
  f.u[1] = *reinterpret_cast<const uint4*>(p + 8);
  return f.v;
}

// NTM head addressing: cosine softmax -> gate blend -> circular shift -> sharpen.
// prpw: LDS [BT][PRW]; off selects head (0=read, 80=write). prev_w/w_out may alias.
__device__ void address_head(const float* prpw, int off, const float* memf,
                             const float* prev_w, float* w_out,
                             float* wtmp, float* wtmp2,
                             float* kn, float* bt, float* gg, float* sh, float* gm,
                             float* mx, float* zz, int tid) {
  if (tid < BT) {
    int b = tid;
    const float* row = prpw + b * PRW + off;
    float k2 = 0.f;
    for (int c = 0; c < C_; ++c) { float v = row[c] + 1e-16f; k2 += v * v; }
    kn[b] = sqrtf(k2);
    bt[b] = softplusf(row[C_]);
    gg[b] = sigm(row[C_ + 1]);
    float a0 = row[C_ + 2], a1 = row[C_ + 3], a2 = row[C_ + 4];
    float m  = fmaxf(a0, fmaxf(a1, a2));
    float e0 = __expf(a0 - m), e1 = __expf(a1 - m), e2 = __expf(a2 - m);
    float es = e0 + e1 + e2;
    sh[b * 3 + 0] = e0 / es; sh[b * 3 + 1] = e1 / es; sh[b * 3 + 2] = e2 / es;
    gm[b] = 1.f + softplusf(row[C_ + 5]);
  }
  __syncthreads();
  // beta * cosine similarity per (b, n)
  for (int i = tid; i < BT * N_; i += 256) {
    int b = i >> 7, n = i & (N_ - 1);
    const float* mrow = memf + ((size_t)b * N_ + n) * C_;
    const float* key  = prpw + b * PRW + off;
    float num = 0.f, na = 0.f;
    for (int c = 0; c < C_; ++c) {
      float a = mrow[c] + 1e-16f, k = key[c] + 1e-16f;
      num += a * k; na += a * a;
    }
    float den = fmaxf(sqrtf(na) * kn[b], 1e-8f);
    wtmp[i] = bt[b] * (num / den);
  }
  __syncthreads();
  if (tid < BT) {
    int b = tid; float m = -1e30f;
    for (int n = 0; n < N_; ++n) m = fmaxf(m, wtmp[b * N_ + n]);
    float Z = 0.f;
    for (int n = 0; n < N_; ++n) Z += __expf(wtmp[b * N_ + n] - m);
    mx[b] = m; zz[b] = Z;
  }
  __syncthreads();
  for (int i = tid; i < BT * N_; i += 256) {
    int b = i >> 7;
    float wc = __expf(wtmp[i] - mx[b]) / zz[b];
    wtmp2[i] = gg[b] * wc + (1.f - gg[b]) * prev_w[i];
  }
  __syncthreads();
  for (int i = tid; i < BT * N_; i += 256) {
    int b = i >> 7, n = i & (N_ - 1);
    int nm = (n + N_ - 1) & (N_ - 1), np = (n + 1) & (N_ - 1);
    float wv = sh[b * 3 + 0] * wtmp2[b * N_ + nm]
             + sh[b * 3 + 1] * wtmp2[i]
             + sh[b * 3 + 2] * wtmp2[b * N_ + np];
    wtmp[i] = __powf(fmaxf(wv, 0.f), gm[b]);
  }
  __syncthreads();
  if (tid < BT) {
    int b = tid; float Z = 0.f;
    for (int n = 0; n < N_; ++n) Z += wtmp[b * N_ + n];
    zz[b] = Z + 1e-16f;
  }
  __syncthreads();
  for (int i = tid; i < BT * N_; i += 256) w_out[i] = wtmp[i] / zz[i >> 7];
  __syncthreads();
}

// ------------------------- weight prep (f32 -> bf16, padded) ----------------
__global__ void ntm_prep(const float* Wih_f, const float* Whh_f, const float* Wread,
                         const float* Wwrite, const float* Wout_f,
                         const float* bih, const float* bhh, const float* bread,
                         const float* bwrite, const float* bout,
                         bf16_t* Wih, bf16_t* Whh, bf16_t* Wrw, bf16_t* Wout,
                         float* biasg, float* biasrw, float* biasout) {
  const size_t n1 = (size_t)NG * LIN;            // 131072
  const size_t n2 = n1 + (size_t)NG * H_;        // +262144
  const size_t n3 = n2 + (size_t)PRW * H_;       // +73728
  const size_t n4 = n3 + (size_t)OUT_ * KO;      // +20480
  const size_t n5 = n4 + NG;                     // +1024
  const size_t n6 = n5 + PRW;                    // +288
  const size_t n7 = n6 + OUT_;                   // +64
  for (size_t i = (size_t)blockIdx.x * blockDim.x + threadIdx.x; i < n7;
       i += (size_t)gridDim.x * blockDim.x) {
    if (i < n1) { Wih[i] = (bf16_t)Wih_f[i]; }
    else if (i < n2) { size_t j = i - n1; Whh[j] = (bf16_t)Whh_f[j]; }
    else if (i < n3) {
      size_t j = i - n2; int r = (int)(j / H_), k = (int)(j % H_);
      float v = 0.f;
      if (r < 70) v = Wread[r * H_ + k];
      else if (r >= 80 && (r - 80) < 198) v = Wwrite[(r - 80) * H_ + k];
      Wrw[j] = (bf16_t)v;
    }
    else if (i < n4) { size_t j = i - n3; Wout[j] = (bf16_t)Wout_f[j]; }
    else if (i < n5) { size_t j = i - n4; biasg[j] = bih[j] + bhh[j]; }
    else if (i < n6) {
      int r = (int)(i - n5); float v = 0.f;
      if (r < 70) v = bread[r];
      else if (r >= 80 && (r - 80) < 198) v = bwrite[r - 80];
      biasrw[r] = v;
    }
    else { size_t j = i - n6; biasout[j] = bout[j]; }
  }
}

// ------------------------------- main kernel --------------------------------
__global__ __launch_bounds__(256, 1)
void ntm_kernel(const float* __restrict__ inputs, const float* __restrict__ mem_seed,
                const float* __restrict__ read_seed, const float* __restrict__ h0,
                const float* __restrict__ c0, float* __restrict__ out,
                float* __restrict__ memf_all,
                const bf16_t* __restrict__ Wih, const bf16_t* __restrict__ Whh,
                const bf16_t* __restrict__ Wrw, const bf16_t* __restrict__ Wout,
                const float* __restrict__ biasg, const float* __restrict__ biasrw,
                const float* __restrict__ biasout) {
  const int tid  = threadIdx.x;
  const int lane = tid & 31;
  const int wv   = tid >> 5;           // wave id, 8 waves
  const int b0   = blockIdx.x * BT;    // batch tile base
  float* memf = memf_all + (size_t)b0 * N_ * C_;

  __shared__ __align__(16) bf16_t s_x[BT * LIN];      // LSTM input [x_t | rd], bf16
  __shared__ __align__(16) bf16_t s_h[BT * H_];       // hidden, bf16 (GEMM A)
  __shared__ __align__(16) float  s_c[BT * H_];       // cell, f32
  __shared__ __align__(16) bf16_t s_rdbf[BT * C_];    // read vector, bf16
  __shared__ __align__(16) float  s_rw[BT * N_];      // read weights
  __shared__ __align__(16) float  s_ww[BT * N_];      // write weights
  __shared__ union ScratchU {
    float gates[BT * NG];                              // 64 KB (phase A only)
    struct { float prpw[BT * PRW]; float wtmp[BT * N_]; float wtmp2[BT * N_]; float er[BT * C_]; } s2;
  } s_u;
  __shared__ float s_kn[16], s_bt[16], s_g[16], s_sh[48], s_gam[16], s_mx[16], s_Z[16];

  // ---- init carry state ----
  for (int i = tid; i < BT * H_; i += 256) {
    int j = i & (H_ - 1);
    s_c[i] = c0[j]; s_h[i] = (bf16_t)h0[j];
  }
  for (int i = tid; i < BT * C_; i += 256) s_rdbf[i] = (bf16_t)read_seed[i & (C_ - 1)];
  for (int i = tid; i < BT * N_; i += 256) { s_rw[i] = 0.f; s_ww[i] = 0.f; }
  for (int i = tid; i < BT * N_ * C_; i += 256) memf[i] = mem_seed[i & (N_ * C_ - 1)];
  __syncthreads();

  for (int t = 0; t < T_; ++t) {
    // ---- build x = [inputs[t] | rd] in bf16 ----
    for (int i = tid; i < BT * LIN; i += 256) {
      int m = i >> 7, j = i & (LIN - 1);
      s_x[i] = (j < IN_) ? (bf16_t)inputs[(size_t)t * B_ * IN_ + (size_t)(b0 + m) * IN_ + j]
                         : s_rdbf[m * C_ + (j - IN_)];
    }
    __syncthreads();

    // ---- phase A: gates = x@Wih^T + h@Whh^T + bias  (64 N-tiles over 8 waves)
    for (int tt = wv; tt < NG / 16; tt += 8) {
      v8f acc;
      float bias = biasg[tt * 16 + (lane & 15)];
      #pragma unroll
      for (int v = 0; v < 8; ++v) acc[v] = bias;
      #pragma unroll
      for (int k0 = 0; k0 < LIN; k0 += 32)
        acc = wmma_bf16(load_a_lds(s_x, LIN, k0, lane), load_b_glob(Wih, LIN, tt * 16, k0, lane), acc);
      #pragma unroll
      for (int k0 = 0; k0 < H_; k0 += 32)
        acc = wmma_bf16(load_a_lds(s_h, H_, k0, lane), load_b_glob(Whh, H_, tt * 16, k0, lane), acc);
      int hi = lane >> 4, n = lane & 15;
      #pragma unroll
      for (int v = 0; v < 8; ++v) s_u.gates[(v + 8 * hi) * NG + tt * 16 + n] = acc[v];
    }
    __syncthreads();

    // ---- LSTM cell elementwise ----
    for (int i = tid; i < BT * H_; i += 256) {
      int m = i >> 8, j = i & (H_ - 1);
      const float* g = s_u.gates + m * NG;
      float iv = g[j], fv = g[H_ + j], gv = g[2 * H_ + j], ov = g[3 * H_ + j];
      float cn = sigm(fv) * s_c[i] + sigm(iv) * tanhf(gv);
      s_c[i] = cn;
      s_h[i] = (bf16_t)(sigm(ov) * tanhf(cn));
    }
    __syncthreads();

    // ---- phase B: [pr | pw] = h @ Wrw^T + bias  (18 N-tiles) ----
    for (int tt = wv; tt < PRW / 16; tt += 8) {
      v8f acc;
      float bias = biasrw[tt * 16 + (lane & 15)];
      #pragma unroll
      for (int v = 0; v < 8; ++v) acc[v] = bias;
      #pragma unroll
      for (int k0 = 0; k0 < H_; k0 += 32)
        acc = wmma_bf16(load_a_lds(s_h, H_, k0, lane), load_b_glob(Wrw, H_, tt * 16, k0, lane), acc);
      int hi = lane >> 4, n = lane & 15;
      #pragma unroll
      for (int v = 0; v < 8; ++v) s_u.s2.prpw[(v + 8 * hi) * PRW + tt * 16 + n] = acc[v];
    }
    __syncthreads();

    // ---- read head addressing (old memory) + read vector ----
    address_head(s_u.s2.prpw, 0, memf, s_rw, s_rw, s_u.s2.wtmp, s_u.s2.wtmp2,
                 s_kn, s_bt, s_g, s_sh, s_gam, s_mx, s_Z, tid);
    for (int i = tid; i < BT * C_; i += 256) {
      int b = i >> 6, c = i & (C_ - 1);
      const float* mb = memf + (size_t)b * N_ * C_ + c;
      float acc = 0.f;
      for (int n = 0; n < N_; ++n) acc += s_rw[b * N_ + n] * mb[n * C_];
      s_rdbf[i] = (bf16_t)acc;
    }

    // ---- write head addressing (old memory) ----
    address_head(s_u.s2.prpw, 80, memf, s_ww, s_ww, s_u.s2.wtmp, s_u.s2.wtmp2,
                 s_kn, s_bt, s_g, s_sh, s_gam, s_mx, s_Z, tid);

    // ---- erase/add memory update ----
    for (int i = tid; i < BT * C_; i += 256) {
      int b = i >> 6, c = i & (C_ - 1);
      s_u.s2.er[i] = sigm(s_u.s2.prpw[b * PRW + 80 + 70 + c]);   // sigmoid(erase)
    }
    __syncthreads();
    for (int i = tid; i < BT * N_ * C_; i += 256) {
      int b = i >> 13, n = (i >> 6) & (N_ - 1), c = i & (C_ - 1);
      float wwv = s_ww[b * N_ + n];
      float e   = s_u.s2.er[b * C_ + c];
      float ad  = s_u.s2.prpw[b * PRW + 80 + 134 + c];            // add (raw)
      memf[i] = memf[i] * (1.f - wwv * e) + wwv * ad;
    }

    // ---- phase E: out = sigmoid([h | rd] @ Wout^T + b_out)  (4 tiles, waves 0-3)
    if (wv < 4) {
      int tt = wv;
      v8f acc;
      float bias = biasout[tt * 16 + (lane & 15)];
      #pragma unroll
      for (int v = 0; v < 8; ++v) acc[v] = bias;
      #pragma unroll
      for (int kt = 0; kt < KO / 32; ++kt) {
        int k0 = kt * 32;
        v16bf a = (k0 < H_) ? load_a_lds(s_h, H_, k0, lane)
                            : load_a_lds(s_rdbf, C_, k0 - H_, lane);
        acc = wmma_bf16(a, load_b_glob(Wout, KO, tt * 16, k0, lane), acc);
      }
      int hi = lane >> 4, n = lane & 15;
      #pragma unroll
      for (int v = 0; v < 8; ++v) {
        int m = v + 8 * hi;
        out[(size_t)t * B_ * OUT_ + (size_t)(b0 + m) * OUT_ + tt * 16 + n] = sigm(acc[v]);
      }
    }
    __syncthreads();
  }
}

// ------------------------------- launcher -----------------------------------
extern "C" void kernel_launch(void* const* d_in, const int* in_sizes, int n_in,
                              void* d_out, int out_size, void* d_ws, size_t ws_size,
                              hipStream_t stream) {
  (void)in_sizes; (void)n_in; (void)out_size; (void)ws_size;
  const float* inputs    = (const float*)d_in[0];
  const float* mem_seed  = (const float*)d_in[1];
  const float* read_seed = (const float*)d_in[2];
  const float* h0        = (const float*)d_in[3];
  const float* c0        = (const float*)d_in[4];
  const float* W_ih      = (const float*)d_in[5];
  const float* W_hh      = (const float*)d_in[6];
  const float* b_ih      = (const float*)d_in[7];
  const float* b_hh      = (const float*)d_in[8];
  const float* W_read    = (const float*)d_in[9];
  const float* b_read    = (const float*)d_in[10];
  const float* W_write   = (const float*)d_in[11];
  const float* b_write   = (const float*)d_in[12];
  const float* W_out     = (const float*)d_in[13];
  const float* b_out     = (const float*)d_in[14];

  char* ws = (char*)d_ws;
  float*  memf    = (float*)(ws + OFF_MEM);
  bf16_t* Wih     = (bf16_t*)(ws + OFF_WIH);
  bf16_t* Whh     = (bf16_t*)(ws + OFF_WHH);
  bf16_t* Wrw     = (bf16_t*)(ws + OFF_WRW);
  bf16_t* Woutb   = (bf16_t*)(ws + OFF_WOUT);
  float*  biasg   = (float*)(ws + OFF_BG);
  float*  biasrw  = (float*)(ws + OFF_BRW);
  float*  biasout = (float*)(ws + OFF_BOUT);

  ntm_prep<<<dim3(1910), dim3(256), 0, stream>>>(
      W_ih, W_hh, W_read, W_write, W_out, b_ih, b_hh, b_read, b_write, b_out,
      Wih, Whh, Wrw, Woutb, biasg, biasrw, biasout);

  ntm_kernel<<<dim3(B_ / BT), dim3(256), 0, stream>>>(
      inputs, mem_seed, read_seed, h0, c0, (float*)d_out, memf,
      Wih, Whh, Wrw, Woutb, biasg, biasrw, biasout);
}
